// TGNN_degree_guided_40673340293689
// MI455X (gfx1250) — matrix operand order repaired and checked by reading, not compiled, not run on hardware
//
#include <hip/hip_runtime.h>
#include <math.h>

// ---------------------------------------------------------------------------
// Problem constants (from the reference)
// ---------------------------------------------------------------------------
#define NN     20000      // nodes (multiple of 16)
#define BBATCH 20         // graphs
#define ET     128000     // symmetric message-passing edges (2*E_HALF)
#define EFULL  400000
#define EACT   200000
#define NACT   10000
#define ECHUNK 20000      // active-edge chunk (multiple of 16; 10 chunks)
#define KMAX   512        // largest GEMM K in this network

typedef __attribute__((ext_vector_type(16))) _Float16 v16h;
typedef __attribute__((ext_vector_type(8)))  _Float16 half8;
typedef __attribute__((ext_vector_type(4)))  _Float16 half4;
typedef __attribute__((ext_vector_type(8)))  float    v8f;

// ---------------------------------------------------------------------------
// WMMA GEMM:  C[M,N] = act(A[M,K] @ W[K,N] + bias)
//   A  : f32 row-major [M,K], K multiple of 32, rows readable to 16-row tile end
//   Wh : f16 TRANSPOSED [N,K] (pre-converted), N multiple of 16
// Block = 8 waves sharing one 16-row A tile. The A panel (16 x K) is staged
// once per block into LDS as f16 (coalesced b128 global loads + packed cvt),
// with a +8-half row pad to spread the per-lane ds_load_b128 across banks.
// K loop body: 2x ds_load_b128 (A frag) + 2x global_load_b128 (f16 B frag)
// + v_wmma_f32_16x16x32_f16. No guards, no converts in the loop.
// ---------------------------------------------------------------------------
#define LDS_STRIDE (KMAX + 8)

template <int ACT>
__global__ __launch_bounds__(256) void k_gemm(const float* __restrict__ A,
                                              const _Float16* __restrict__ Wh,
                                              const float* __restrict__ bias,
                                              float* __restrict__ C,
                                              int M, int K, int Ncols, int ldC) {
  __shared__ __align__(16) _Float16 lds[16 * LDS_STRIDE];
  const int wave = threadIdx.x >> 5;
  const int lane = threadIdx.x & 31;
  const int tm = blockIdx.x;
  const int tn = blockIdx.y * 8 + wave;
  const bool active = (tn * 16 < Ncols);
  const int grp = lane >> 4;                 // half-wave group
  const int ml  = lane & 15;
  const int ldsStride = K + 8;               // halves; row skew vs 64x4B banks

  // --- cooperative A panel stage: 16 rows x K f32 -> f16 in LDS ---
  {
    const int kq = K >> 2;                   // float4s per row
    const int nf4 = kq * 16;
    const float4* __restrict__ a4 = (const float4*)A;
    for (int g = threadIdx.x; g < nf4; g += 256) {
      int row = g / kq;
      int kc4 = g - row * kq;
      float4 a = a4[(size_t)(tm * 16 + row) * kq + kc4];
      half4 hv;
      hv[0] = (_Float16)a.x; hv[1] = (_Float16)a.y;
      hv[2] = (_Float16)a.z; hv[3] = (_Float16)a.w;
      *(half4*)(lds + row * ldsStride + (kc4 << 2)) = hv;
    }
  }
  __syncthreads();

  if (active) {
    const int bcol = tn * 16 + ml;           // B column owned by this lane
    const half8* __restrict__ lA = (const half8*)(lds + ml * ldsStride);
    const half8* __restrict__ bp = (const half8*)(Wh + (size_t)bcol * K + grp * 8);
    v8f acc = {};
    for (int k0 = 0; k0 < K; k0 += 32) {
      half8 a0 = lA[(k0 >> 3) + grp];        // halves k0 + grp*8 .. +7
      half8 a1 = lA[(k0 >> 3) + 2 + grp];    // halves k0 + 16 + grp*8 .. +7
      half8 b0 = bp[(k0 >> 3)];
      half8 b1 = bp[(k0 >> 3) + 2];
      v16h av, bv;
#pragma unroll
      for (int j = 0; j < 8; ++j) {
        av[j] = a0[j]; av[j + 8] = a1[j];
        bv[j] = b0[j]; bv[j + 8] = b1[j];
      }
      acc = __builtin_amdgcn_wmma_f32_16x16x32_f16(false, av, false, bv,
                                                   (short)0, acc, false, false);
    }
    const float bb = bias[bcol];
#pragma unroll
    for (int r = 0; r < 8; ++r) {            // C/D: VGPR r -> M = r + grp*8
      int orow = tm * 16 + grp * 8 + r;
      if (orow < M) {
        float v = acc[r] + bb;
        if (ACT == 1) v = v / (1.0f + __expf(-v));   // SiLU
        C[(size_t)orow * ldC + bcol] = v;
      }
    }
  }
}

// ---------------------------------------------------------------------------
// Weight pre-conversion: Wh[n*K+k] = (f16) W[k*N+n]   (transpose + cvt)
// ---------------------------------------------------------------------------
#define GIDX size_t idx = (size_t)blockIdx.x * blockDim.x + threadIdx.x; if (idx >= total) return;

__global__ void k_wcvt(const float* __restrict__ W, _Float16* Wh, int K, int N,
                       size_t total) {
  GIDX
  size_t n = idx / (size_t)K; int k = (int)(idx - n * K);
  Wh[idx] = (_Float16)W[(size_t)k * N + n];
}
__global__ void k_wcvt_pad(const float* __restrict__ W, _Float16* Wh, int K,
                           int Nsrc, size_t total) {   // total = Npad*K
  GIDX
  size_t n = idx / (size_t)K; int k = (int)(idx - n * K);
  Wh[idx] = (n < (size_t)Nsrc) ? (_Float16)W[(size_t)k * Nsrc + n] : (_Float16)0.0f;
}
__global__ void k_bias_pad(const float* __restrict__ b, float* bp, int Nsrc,
                           size_t total) {
  GIDX bp[idx] = (idx < (size_t)Nsrc) ? b[idx] : 0.0f;
}

// ---------------------------------------------------------------------------
// Elementwise / graph kernels
// ---------------------------------------------------------------------------
__device__ __forceinline__ float d_sig(float x) { return 1.0f / (1.0f + __expf(-x)); }
__device__ __forceinline__ unsigned f2ord(float f) {
  unsigned u = __float_as_uint(f);
  return (u & 0x80000000u) ? ~u : (u | 0x80000000u);
}
__device__ __forceinline__ float ord2f(unsigned u) {
  return (u & 0x80000000u) ? __uint_as_float(u & 0x7fffffffu) : __uint_as_float(~u);
}

__global__ void k_zero_f(float* p, size_t total) { GIDX p[idx] = 0.0f; }
__global__ void k_zero_i(int* p, size_t total)   { GIDX p[idx] = 0; }

__global__ void k_deg(const int* __restrict__ ei, float* deg, size_t total) {
  GIDX atomicAdd(&deg[ei[idx]], 1.0f);   // src row of edge_index_t
}
__global__ void k_sel(const int* __restrict__ an, int* sel, size_t total) {
  GIDX sel[an[idx]] = 1;
}
__global__ void k_node_embed(const float* __restrict__ deg, const float* __restrict__ degree,
                             const int* __restrict__ sel,
                             const float* Wt, const float* bt,
                             const float* W0, const float* b0,
                             const float* EmbSel,
                             float* e_t, float* e_0, float* e_s, float* cat3,
                             size_t total) {
  GIDX
  size_t n = idx >> 7; int d = idx & 127;
  float dt = fminf(floorf(deg[n]), 101.0f) * 0.01f;
  float d0 = degree[n] * 0.01f;
  float et = dt * Wt[d] + bt[d];
  float e0 = d0 * W0[d] + b0[d];
  float es = EmbSel[(size_t)sel[n] * 128 + d];
  e_t[idx] = et; e_0[idx] = e0; e_s[idx] = es;
  cat3[n * 384 + d] = et; cat3[n * 384 + 128 + d] = e0; cat3[n * 384 + 256 + d] = es;
}
__global__ void k_time_pe(const float* __restrict__ t_node, float* pe, size_t total) {
  GIDX
  size_t n = idx >> 6; int j = idx & 63;
  float x = t_node[n] * (4000.0f / 512.0f);
  float a = x * __expf(-0.14619588f * (float)j);   // -ln(10000)/63
  pe[n * 128 + j] = __sinf(a);
  pe[n * 128 + 64 + j] = __cosf(a);
}
__global__ void k_mish(const float* __restrict__ in, float* out, size_t total) {
  GIDX
  float x = in[idx];
  float sp = (x > 20.0f) ? x : log1pf(__expf(x));
  out[idx] = x * tanhf(sp);
}
__global__ void k_place(const float* __restrict__ src, int sc, float* dst,
                        int ld, int off, size_t total) {
  GIDX
  size_t r = idx / (size_t)sc; int c = (int)(idx - r * sc);
  dst[r * (size_t)ld + off + c] = src[idx];
}
__global__ void k_add(float* a, const float* __restrict__ b, size_t total) { GIDX a[idx] += b[idx]; }
__global__ void k_counts(const int* __restrict__ batch, float* counts, size_t total) {
  GIDX atomicAdd(&counts[batch[idx]], 1.0f);
}
__global__ void k_segsum(const float* __restrict__ x, const int* __restrict__ batch,
                         float* g, size_t total) {
  GIDX size_t n = idx >> 7; atomicAdd(&g[(size_t)batch[n] * 128 + (idx & 127)], x[idx]);
}
__global__ void k_segsum2(const float* __restrict__ x, const float* __restrict__ y,
                          const int* __restrict__ batch, float* g, size_t total) {
  GIDX size_t n = idx >> 7; atomicAdd(&g[(size_t)batch[n] * 128 + (idx & 127)], x[idx] + y[idx]);
}
__global__ void k_div(float* g, const float* __restrict__ counts, size_t total) {
  GIDX g[idx] /= counts[idx >> 7];
}
__global__ void k_gather(const int* __restrict__ batch, const float* __restrict__ g,
                         float* ctx, size_t total) {
  GIDX size_t n = idx >> 7; ctx[idx] = g[(size_t)batch[n] * 128 + (idx & 127)];
}
__global__ void k_attn_scores(const int* __restrict__ ei, const float* __restrict__ q,
                              const float* __restrict__ k, float* alpha,
                              int H, float scale, size_t total) {
  GIDX
  size_t e = idx / H; int h = (int)(idx - e * H);
  int s = ei[e], d = ei[ET + e];
  const float4* qp = (const float4*)(q + (size_t)d * H * 128 + (size_t)h * 128);
  const float4* kp = (const float4*)(k + (size_t)s * H * 128 + (size_t)h * 128);
  float acc = 0.0f;
#pragma unroll 8
  for (int i = 0; i < 32; ++i) {
    float4 a = qp[i], b = kp[i];
    acc += a.x * b.x + a.y * b.y + a.z * b.z + a.w * b.w;
  }
  alpha[idx] = acc * scale;
}
__global__ void k_seg_max(const int* __restrict__ ei, const float* __restrict__ alpha,
                          unsigned* amax, int H, size_t total) {
  GIDX
  size_t e = idx / H; int h = (int)(idx - e * H);
  atomicMax(&amax[(size_t)ei[ET + e] * H + h], f2ord(alpha[idx]));
}
__global__ void k_attn_exp(const int* __restrict__ ei, float* alpha,
                           const unsigned* __restrict__ amax, float* denom,
                           int H, size_t total) {
  GIDX
  size_t e = idx / H; int h = (int)(idx - e * H);
  int d = ei[ET + e];
  float ae = __expf(alpha[idx] - ord2f(amax[(size_t)d * H + h]));
  alpha[idx] = ae;
  atomicAdd(&denom[(size_t)d * H + h], ae);
}
__global__ void k_attn_agg(const int* __restrict__ ei, const float* __restrict__ ae,
                           const float* __restrict__ denom, const float* __restrict__ vbuf,
                           float* agg, int H, size_t total) {
  GIDX
  int HC = H * 128;
  size_t e = idx / HC; int rem = (int)(idx - e * HC); int h = rem >> 7;
  int s = ei[e], d = ei[ET + e];
  float w = ae[e * H + h] / denom[(size_t)d * H + h];
  atomicAdd(&agg[(size_t)d * HC + rem], w * vbuf[(size_t)s * HC + rem]);
}
__global__ void k_headmean(const float* __restrict__ agg, const float* __restrict__ skip,
                           float* nodes, int H, size_t total) {
  GIDX
  size_t n = idx >> 7; int d = idx & 127;
  float s = 0.0f;
  for (int h = 0; h < H; ++h) s += agg[(size_t)n * H * 128 + (size_t)h * 128 + d];
  float v = s / (float)H + skip[idx];
  nodes[idx] = v / (1.0f + __expf(-v));
}
__global__ void k_gru(const float* __restrict__ gx, const float* __restrict__ gh,
                      float* h, float* nodes, size_t total) {
  GIDX
  size_t n = idx >> 7; int d = idx & 127;
  const float* gxp = gx + n * 384; const float* ghp = gh + n * 384;
  float r  = d_sig(gxp[d] + ghp[d]);
  float z  = d_sig(gxp[128 + d] + ghp[128 + d]);
  float ng = tanhf(gxp[256 + d] + r * ghp[256 + d]);
  float hn = (1.0f - z) * ng + z * h[idx];
  h[idx] = hn; nodes[idx] = hn;
}
__global__ void k_edge_gather(const int* __restrict__ full_ei, const int* __restrict__ acte,
                              const float* __restrict__ nf, float* ee,
                              int base, size_t total) {  // total = cnt*64 (float4)
  GIDX
  size_t i = idx >> 6; int d4 = idx & 63;
  int j = acte[base + i];
  int r = full_ei[j], c = full_ei[EFULL + j];
  const float4* pr = (const float4*)(nf + (size_t)r * 256);
  const float4* pc = (const float4*)(nf + (size_t)c * 256);
  float4 a = pr[d4], b = pc[d4];
  float4 o; o.x = a.x + b.x; o.y = a.y + b.y; o.z = a.z + b.z; o.w = a.w + b.w;
  ((float4*)ee)[idx] = o;
}
__global__ void k_take2(const float* __restrict__ ein, float* out, size_t total) {
  GIDX size_t i = idx >> 1; out[idx] = ein[i * 16 + (idx & 1)];
}

// ---------------------------------------------------------------------------
// Host side
// ---------------------------------------------------------------------------
static inline dim3 G(size_t n) { return dim3((unsigned)((n + 255) / 256)); }

static void gemm(hipStream_t s, const float* A, const _Float16* Wh, const float* b,
                 float* C, int M, int K, int N, int ldC, int act) {
  dim3 g((M + 15) / 16, (N + 127) / 128);
  if (act) k_gemm<1><<<g, 256, 0, s>>>(A, Wh, b, C, M, K, N, ldC);
  else     k_gemm<0><<<g, 256, 0, s>>>(A, Wh, b, C, M, K, N, ldC);
}
static void wcvt(hipStream_t s, const float* W, _Float16* Wh, int K, int N) {
  size_t t = (size_t)K * N;
  k_wcvt<<<G(t), 256, 0, s>>>(W, Wh, K, N, t);
}

extern "C" void kernel_launch(void* const* d_in, const int* in_sizes, int n_in,
                              void* d_out, int out_size, void* d_ws, size_t ws_size,
                              hipStream_t stream) {
  (void)in_sizes; (void)n_in; (void)out_size; (void)ws_size;
  const int*   edge_t  = (const int*)d_in[0];
  const float* degree  = (const float*)d_in[1];
  const int*   actn    = (const int*)d_in[2];
  const int*   batch   = (const int*)d_in[3];
  const int*   full_ei = (const int*)d_in[5];
  const int*   acte    = (const int*)d_in[6];
  const float* lna     = (const float*)d_in[7];
  const float* t_node  = (const float*)d_in[8];
  // params flattened in jax pytree (sorted-key) order after the 10 inputs
  const float* P[75];
  for (int i = 0; i < 75; ++i) P[i] = (const float*)d_in[10 + i];
  const float *Wc1=P[0], *bc1=P[1], *Wc2=P[2], *bc2=P[3];           // context_mlp
  const float *W0e=P[4], *b0e=P[5];                                  // emb_0
  const float *EmbSel=P[6];                                          // emb_sel
  const float *Wte=P[7], *bte=P[8];                                  // emb_t
  const float *Wf1=P[9], *bf1=P[10], *Wf2=P[11], *bf2=P[12], *Wf3=P[13], *bf3=P[14]; // final
  const float *Wg1=P[15], *bg1=P[16], *Wg2=P[17], *bg2=P[18];        // global_mlp
  const float *Wih=P[19], *Whh=P[20], *bih=P[21], *bhh=P[22];        // gru
  const float *Wm1=P[63], *bm1=P[64], *Wm2=P[65], *bm2=P[66];        // mlp
  const float *Wni=P[67], *bni=P[68];                                // node_in
  const float *Wn1=P[69], *bn1=P[70], *Wn2=P[71], *bn2=P[72], *Wn3=P[73], *bn3=P[74]; // node_out

  // workspace bump allocation — identical layout every call
  float* w = (float*)d_ws; size_t off = 0;
  auto A  = [&](size_t n) { float* p = w + off; off += (n + 3) & ~(size_t)3; return p; };
  auto AH = [&](size_t n) { _Float16* p = (_Float16*)(w + off);
                            off += ((n + 1) / 2 + 3) & ~(size_t)3; return p; };
  float* deg  = A(NN);
  int*   sel  = (int*)A(NN);
  float* e_t  = A((size_t)NN * 128);
  float* e_0  = A((size_t)NN * 128);
  float* e_s  = A((size_t)NN * 128);
  float* cat3 = A((size_t)NN * 384);
  float* tE   = A((size_t)NN * 128);
  float* nodes= A((size_t)NN * 128);
  float* hb   = A((size_t)NN * 128);
  float* ctx  = A((size_t)NN * 128);
  float* xin  = A((size_t)NN * 256);
  float* S0   = A((size_t)NN * 512);
  float* qb   = A((size_t)NN * 512);
  float* kb   = A((size_t)NN * 512);
  float* vb   = A((size_t)NN * 512);
  float* T0   = A((size_t)NN * 128);
  float* T1   = A((size_t)NN * 128);
  float* gx   = A((size_t)NN * 384);
  float* gh   = A((size_t)NN * 384);
  float* alpha= A((size_t)ET * 4);
  unsigned* amax = (unsigned*)A((size_t)NN * 4);
  float* denom= A((size_t)NN * 4);
  float* agg  = A((size_t)NN * 512);
  float* gsum = A((size_t)BBATCH * 128);
  float* gh4  = A((size_t)BBATCH * 512);
  float* gout = A((size_t)BBATCH * 128);
  float* counts = A(BBATCH);
  float* nf   = A((size_t)NN * 256);
  float* EE0  = A((size_t)ECHUNK * 256);
  float* EE1  = A((size_t)ECHUNK * 256);
  float* EE2  = A((size_t)ECHUNK * 128);
  float* EEo  = A((size_t)ECHUNK * 16);
  float* bf3p = A(16);
  // transposed f16 weights
  _Float16* WniT = AH(128 * 384);
  _Float16* Wm1T = AH(512 * 128);
  _Float16* Wm2T = AH(128 * 512);
  _Float16* Wg1T = AH(512 * 128);
  _Float16* Wg2T = AH(128 * 512);
  _Float16* Wc1T = AH(512 * 256);
  _Float16* Wc2T = AH(128 * 512);
  _Float16* WihT = AH(384 * 128);
  _Float16* WhhT = AH(384 * 128);
  _Float16* Wn1T = AH(256 * 512);
  _Float16* Wn2T = AH(256 * 256);
  _Float16* Wn3T = AH(256 * 256);
  _Float16* Wf1T = AH(256 * 256);
  _Float16* Wf2T = AH(128 * 256);
  _Float16* Wf3T = AH(16 * 128);
  _Float16* LqT[4]; _Float16* LkT[4]; _Float16* LvT[4];
  _Float16* LsT[4]; _Float16* LtT[4];
  const int heads[4] = {4, 4, 4, 1};
  for (int L = 0; L < 4; ++L) {
    LkT[L] = AH((size_t)heads[L] * 128 * 256);
    LqT[L] = AH((size_t)heads[L] * 128 * 256);
    LsT[L] = AH(128 * 256);
    LtT[L] = AH(128 * 128);
    LvT[L] = AH((size_t)heads[L] * 128 * 256);
  }

  // --- weight conversion prepass (all L2-resident, tiny) ---
  wcvt(stream, Wni, WniT, 384, 128);
  wcvt(stream, Wm1, Wm1T, 128, 512);
  wcvt(stream, Wm2, Wm2T, 512, 128);
  wcvt(stream, Wg1, Wg1T, 128, 512);
  wcvt(stream, Wg2, Wg2T, 512, 128);
  wcvt(stream, Wc1, Wc1T, 256, 512);
  wcvt(stream, Wc2, Wc2T, 512, 128);
  wcvt(stream, Wih, WihT, 128, 384);
  wcvt(stream, Whh, WhhT, 128, 384);
  wcvt(stream, Wn1, Wn1T, 512, 256);
  wcvt(stream, Wn2, Wn2T, 256, 256);
  wcvt(stream, Wn3, Wn3T, 256, 256);
  wcvt(stream, Wf1, Wf1T, 256, 256);
  wcvt(stream, Wf2, Wf2T, 256, 128);
  k_wcvt_pad<<<G(16 * 128), 256, 0, stream>>>(Wf3, Wf3T, 128, 2, (size_t)16 * 128);
  k_bias_pad<<<G(16), 256, 0, stream>>>(bf3, bf3p, 2, 16);
  for (int L = 0; L < 4; ++L) {
    const int base = 23 + 10 * L;   // layers dict sorted: k,q,skip,time,v
    wcvt(stream, P[base + 0], LkT[L], 256, heads[L] * 128);
    wcvt(stream, P[base + 2], LqT[L], 256, heads[L] * 128);
    wcvt(stream, P[base + 4], LsT[L], 256, 128);
    wcvt(stream, P[base + 6], LtT[L], 128, 128);
    wcvt(stream, P[base + 8], LvT[L], 256, heads[L] * 128);
  }

  const size_t NC = (size_t)NN * 128;

  // --- degree, selection, embeddings, node_in ---
  k_zero_f<<<G(NN), 256, 0, stream>>>(deg, NN);
  k_zero_i<<<G(NN), 256, 0, stream>>>(sel, NN);
  k_deg<<<G(ET), 256, 0, stream>>>(edge_t, deg, ET);
  k_sel<<<G(NACT), 256, 0, stream>>>(actn, sel, NACT);
  k_node_embed<<<G(NC), 256, 0, stream>>>(deg, degree, sel, Wte, bte, W0e, b0e,
                                          EmbSel, e_t, e_0, e_s, cat3, NC);
  gemm(stream, cat3, WniT, bni, nodes, NN, 384, 128, 128, 1);

  // --- sinusoidal time embedding + MLP ---
  k_time_pe<<<G((size_t)NN * 64), 256, 0, stream>>>(t_node, T0, (size_t)NN * 64);
  gemm(stream, T0, Wm1T, bm1, S0, NN, 128, 512, 512, 1);
  gemm(stream, S0, Wm2T, bm2, tE, NN, 512, 128, 128, 0);

  // --- GRU hidden init, counts, initial context ---
  k_place<<<G(NC), 256, 0, stream>>>(nodes, 128, hb, 128, 0, NC);
  k_zero_f<<<G(BBATCH), 256, 0, stream>>>(counts, BBATCH);
  k_counts<<<G(NN), 256, 0, stream>>>(batch, counts, NN);
  k_zero_f<<<G((size_t)BBATCH * 128), 256, 0, stream>>>(gsum, (size_t)BBATCH * 128);
  k_segsum<<<G(NC), 256, 0, stream>>>(nodes, batch, gsum, NC);
  k_div<<<G((size_t)BBATCH * 128), 256, 0, stream>>>(gsum, counts, (size_t)BBATCH * 128);
  gemm(stream, gsum, Wg1T, bg1, gh4, BBATCH, 128, 512, 512, 1);
  gemm(stream, gh4, Wg2T, bg2, gout, BBATCH, 512, 128, 128, 0);
  k_gather<<<G(NC), 256, 0, stream>>>(batch, gout, ctx, NC);

  // --- attention layers ---
  for (int L = 0; L < 4; ++L) {
    const int H = heads[L];
    const int base = 23 + 10 * L;
    const float *lkb = P[base + 1], *lqb = P[base + 3], *lsb = P[base + 5];
    const float *ltb = P[base + 7], *lvb = P[base + 9];

    k_mish<<<G(NC), 256, 0, stream>>>(tE, T0, NC);
    gemm(stream, T0, LtT[L], ltb, xin + 128, NN, 128, 128, 256, 0); // t_emb -> xin[:,128:]
    k_place<<<G(NC), 256, 0, stream>>>(nodes, 128, xin, 256, 0, NC);
    gemm(stream, xin, LqT[L], lqb, qb, NN, 256, H * 128, H * 128, 0);
    gemm(stream, xin, LkT[L], lkb, kb, NN, 256, H * 128, H * 128, 0);
    gemm(stream, xin, LvT[L], lvb, vb, NN, 256, H * 128, H * 128, 0);

    const size_t EH = (size_t)ET * H;
    k_zero_f<<<G((size_t)NN * H), 256, 0, stream>>>((float*)amax, (size_t)NN * H);
    k_zero_f<<<G((size_t)NN * H), 256, 0, stream>>>(denom, (size_t)NN * H);
    k_zero_f<<<G((size_t)NN * H * 128), 256, 0, stream>>>(agg, (size_t)NN * H * 128);
    k_attn_scores<<<G(EH), 256, 0, stream>>>(edge_t, qb, kb, alpha, H,
                                             0.08838834764831845f, EH);
    k_seg_max<<<G(EH), 256, 0, stream>>>(edge_t, alpha, amax, H, EH);
    k_attn_exp<<<G(EH), 256, 0, stream>>>(edge_t, alpha, amax, denom, H, EH);
    k_attn_agg<<<G(EH * 128), 256, 0, stream>>>(edge_t, alpha, denom, vb, agg, H, EH * 128);
    gemm(stream, xin, LsT[L], lsb, T0, NN, 256, 128, 128, 0);       // skip
    k_headmean<<<G(NC), 256, 0, stream>>>(agg, T0, nodes, H, NC);

    // GRU (single step)
    gemm(stream, nodes, WihT, bih, gx, NN, 128, 384, 384, 0);
    gemm(stream, hb, WhhT, bhh, gh, NN, 128, 384, 384, 0);
    k_gru<<<G(NC), 256, 0, stream>>>(gx, gh, hb, nodes, NC);

    // context update
    k_place<<<G(NC), 256, 0, stream>>>(nodes, 128, xin, 256, 0, NC);
    k_place<<<G(NC), 256, 0, stream>>>(ctx, 128, xin, 256, 128, NC);
    gemm(stream, xin, Wc1T, bc1, S0, NN, 256, 512, 512, 1);
    gemm(stream, S0, Wc2T, bc2, T1, NN, 512, 128, 128, 0);          // node_ctx
    k_zero_f<<<G((size_t)BBATCH * 128), 256, 0, stream>>>(gsum, (size_t)BBATCH * 128);
    k_segsum2<<<G(NC), 256, 0, stream>>>(ctx, T1, batch, gsum, NC);
    k_div<<<G((size_t)BBATCH * 128), 256, 0, stream>>>(gsum, counts, (size_t)BBATCH * 128);
    gemm(stream, gsum, Wg1T, bg1, gh4, BBATCH, 128, 512, 512, 1);
    gemm(stream, gh4, Wg2T, bg2, gout, BBATCH, 512, 128, 128, 0);
    k_gather<<<G(NC), 256, 0, stream>>>(batch, gout, ctx, NC);
    k_add<<<G(NC), 256, 0, stream>>>(nodes, ctx, NC);
  }

  // --- node_out chain: concat [nodes,e_t,e_0,e_s] -> 256 -> 256 -> 256 ---
  k_place<<<G(NC), 256, 0, stream>>>(nodes, 128, S0, 512, 0,   NC);
  k_place<<<G(NC), 256, 0, stream>>>(e_t,   128, S0, 512, 128, NC);
  k_place<<<G(NC), 256, 0, stream>>>(e_0,   128, S0, 512, 256, NC);
  k_place<<<G(NC), 256, 0, stream>>>(e_s,   128, S0, 512, 384, NC);
  gemm(stream, S0, Wn1T, bn1, qb, NN, 512, 256, 256, 1);
  gemm(stream, qb, Wn2T, bn2, kb, NN, 256, 256, 256, 1);
  gemm(stream, kb, Wn3T, bn3, nf, NN, 256, 256, 256, 0);

  // output 0: log_node_attr passthrough
  hipMemcpyAsync(d_out, lna, (size_t)NN * 102 * sizeof(float),
                 hipMemcpyDeviceToDevice, stream);

  // --- final edge classifier, chunked over active edges ---
  for (int base = 0; base < EACT; base += ECHUNK) {
    k_edge_gather<<<G((size_t)ECHUNK * 64), 256, 0, stream>>>(
        full_ei, acte, nf, EE0, base, (size_t)ECHUNK * 64);
    gemm(stream, EE0, Wf1T, bf1, EE1, ECHUNK, 256, 256, 256, 1);
    gemm(stream, EE1, Wf2T, bf2, EE2, ECHUNK, 256, 128, 128, 1);
    gemm(stream, EE2, Wf3T, bf3p, EEo, ECHUNK, 128, 16, 16, 0);
    k_take2<<<G((size_t)ECHUNK * 2), 256, 0, stream>>>(
        EEo, (float*)d_out + (size_t)NN * 102 + (size_t)base * 2,
        (size_t)ECHUNK * 2);
  }
}
